// AFT_Full_48825188221504
// MI455X (gfx1250) — compile-verified, matrix operand order checked by reading, not run on hardware
//
#include <hip/hip_runtime.h>
#include <math.h>

// ---------------------------------------------------------------------------
// AFT-Full, simplified exactly: exp_wb == 1 (max over singleton axis), so
//   num[b,t,h] = sum_s exp_k[b,s,h]*v[b,s,h],  den[b,t,h] = sum_s exp_k[b,s,h]
// Pipeline: qkv bf16-WMMA GEMM -> batch max(k) -> per-(b,h) ratio R ->
//           out = (sigmoid(q)*R) @ Wo + bo  (bf16-WMMA GEMM, fused A-transform)
// ---------------------------------------------------------------------------

typedef __attribute__((ext_vector_type(16))) __bf16 v16bf;
typedef __attribute__((ext_vector_type(8)))  float  v8f;

#define BATCH 64
#define TT    576            // sequence length (24*24)
#define CC    768            // channels
#define HIDD  576            // hidden
#define MROWS (BATCH * TT)   // 36864 rows
#define BM    256
#define BN    64
#define BK    32
#define LDK   56             // padded LDS row stride in halfs (112B: 16B-aligned, conflict-free)

__device__ __forceinline__ unsigned short f2bf(float f) {
  unsigned int u = __float_as_uint(f);
  u += 0x7FFFu + ((u >> 16) & 1u);     // round-to-nearest-even
  return (unsigned short)(u >> 16);
}

union FragU { v16bf v; uint4 q[2]; };

// Load one 16x32 bf16 fragment (A or transposed-B) from LDS per the ISA layout:
// lanes 0-15: halfs 0..7 -> K=0..7, halfs 8..15 -> K=16..23
// lanes16-31: halfs 0..7 -> K=8..15, halfs 8..15 -> K=24..31
__device__ __forceinline__ v16bf load_frag(const unsigned short* base, int row, int hi) {
  FragU f;
  const unsigned short* p = base + row * LDK + hi;
  f.q[0] = *reinterpret_cast<const uint4*>(p);        // K = hi .. hi+7
  f.q[1] = *reinterpret_cast<const uint4*>(p + 16);   // K = hi+16 .. hi+23
  return f.v;
}

// ---------------------------------------------------------------------------
// Kernel 1: q/k/v projections.  grid = (HID/BN, M/BM, 3), block = 256 (8 waves)
// ---------------------------------------------------------------------------
__global__ void __launch_bounds__(256)
aft_gemm_qkv(const float* __restrict__ x,
             const float* __restrict__ Wq, const float* __restrict__ Wk,
             const float* __restrict__ Wv,
             const float* __restrict__ bq, const float* __restrict__ bk,
             const float* __restrict__ bv,
             float* __restrict__ q, float* __restrict__ k, float* __restrict__ v)
{
  __shared__ unsigned short Al[BM * LDK];
  __shared__ unsigned short Bl[BN * LDK];

  const int z = blockIdx.z;
  const float* W  = (z == 0) ? Wq : (z == 1) ? Wk : Wv;
  const float* bs = (z == 0) ? bq : (z == 1) ? bk : bv;
  float*       out = (z == 0) ? q  : (z == 1) ? k  : v;

  const int n0   = blockIdx.x * BN;
  const int m0   = blockIdx.y * BM;
  const int tid  = threadIdx.x;
  const int lane = tid & 31;
  const int wid  = tid >> 5;
  const int wm   = (wid & 3) * 64;     // wave row offset inside block
  const int wn   = (wid >> 2) * 32;    // wave col offset inside block
  const int ln15 = lane & 15;
  const int hi   = (lane >> 4) * 8;
  const int rhi  = (lane >> 4) * 8;    // C/D row offset

  v8f acc[4][2];
  const v8f zero = {0.f, 0.f, 0.f, 0.f, 0.f, 0.f, 0.f, 0.f};
#pragma unroll
  for (int mt = 0; mt < 4; ++mt)
#pragma unroll
    for (int nt = 0; nt < 2; ++nt) acc[mt][nt] = zero;

  for (int kt = 0; kt < CC; kt += BK) {
    __syncthreads();
    // stage A tile (BM x BK fp32 -> bf16), coalesced float4 reads
#pragma unroll
    for (int p = 0; p < 8; ++p) {
      int idx = p * 256 + tid;
      int r = idx >> 3, c4 = (idx & 7) * 4;
      const float4 a = *reinterpret_cast<const float4*>(
          &x[(size_t)(m0 + r) * CC + kt + c4]);
      unsigned long long pk =  (unsigned long long)f2bf(a.x)
        | ((unsigned long long)f2bf(a.y) << 16)
        | ((unsigned long long)f2bf(a.z) << 32)
        | ((unsigned long long)f2bf(a.w) << 48);
      *reinterpret_cast<unsigned long long*>(&Al[r * LDK + c4]) = pk;
    }
    // stage B tile (BK x BN fp32) transposed -> Bl[n][k] bf16
#pragma unroll
    for (int p = 0; p < 2; ++p) {
      int idx = p * 256 + tid;
      int kr = idx >> 4, c4 = (idx & 15) * 4;
      const float4 b = *reinterpret_cast<const float4*>(
          &W[(size_t)(kt + kr) * HIDD + n0 + c4]);
      Bl[(c4 + 0) * LDK + kr] = f2bf(b.x);
      Bl[(c4 + 1) * LDK + kr] = f2bf(b.y);
      Bl[(c4 + 2) * LDK + kr] = f2bf(b.z);
      Bl[(c4 + 3) * LDK + kr] = f2bf(b.w);
    }
    __syncthreads();

    v16bf af[4], bf[2];
#pragma unroll
    for (int mt = 0; mt < 4; ++mt) af[mt] = load_frag(Al, wm + mt * 16 + ln15, hi);
#pragma unroll
    for (int nt = 0; nt < 2; ++nt) bf[nt] = load_frag(Bl, wn + nt * 16 + ln15, hi);
#pragma unroll
    for (int mt = 0; mt < 4; ++mt)
#pragma unroll
      for (int nt = 0; nt < 2; ++nt)
        acc[mt][nt] = __builtin_amdgcn_wmma_f32_16x16x32_bf16(
            false, af[mt], false, bf[nt], (short)0, acc[mt][nt], false, false);
  }

  // epilogue: + bias, fp32 store
#pragma unroll
  for (int mt = 0; mt < 4; ++mt)
#pragma unroll
    for (int nt = 0; nt < 2; ++nt) {
      const int col  = n0 + wn + nt * 16 + ln15;
      const float bb = bs[col];
      const int rb   = m0 + wm + mt * 16 + rhi;
#pragma unroll
      for (int r = 0; r < 8; ++r)
        out[(size_t)(rb + r) * HIDD + col] = acc[mt][nt][r] + bb;
    }
}

// ---------------------------------------------------------------------------
// Kernel 2: maxk[t,h] = max over b of k[b,t,h]
// ---------------------------------------------------------------------------
__global__ void __launch_bounds__(256)
aft_maxk(const float* __restrict__ k, float* __restrict__ mk)
{
  const int i = blockIdx.x * 256 + threadIdx.x;   // i in [0, T*HID)
  float m = k[i];
#pragma unroll 4
  for (int b = 1; b < BATCH; ++b)
    m = fmaxf(m, k[(size_t)b * TT * HIDD + i]);
  mk[i] = m;
}

// ---------------------------------------------------------------------------
// Kernel 3: R[b,h] = sum_t e*v / sum_t e,  e = exp(k[b,t,h]-maxk[t,h])
// ---------------------------------------------------------------------------
__global__ void __launch_bounds__(256)
aft_sums(const float* __restrict__ k, const float* __restrict__ v,
         const float* __restrict__ mk, float* __restrict__ R)
{
  const int i = blockIdx.x * 256 + threadIdx.x;   // i in [0, B*HID)
  const int b = i / HIDD, h = i % HIDD;
  const float* kb = k + (size_t)b * TT * HIDD + h;
  const float* vb = v + (size_t)b * TT * HIDD + h;
  float sn = 0.f, sd = 0.f;
  for (int t = 0; t < TT; ++t) {
    const float e = __expf(kb[(size_t)t * HIDD] - mk[t * HIDD + h]);
    sn += e * vb[(size_t)t * HIDD];
    sd += e;
  }
  R[i] = sn / sd;
}

// ---------------------------------------------------------------------------
// Kernel 4: out = (sigmoid(q) * R[b,:]) @ Wo + bo.  grid = (C/BN, M/BM)
// ---------------------------------------------------------------------------
__global__ void __launch_bounds__(256)
aft_gemm_out(const float* __restrict__ q, const float* __restrict__ R,
             const float* __restrict__ Wo, const float* __restrict__ bo,
             float* __restrict__ out)
{
  __shared__ unsigned short Al[BM * LDK];
  __shared__ unsigned short Bl[BN * LDK];

  const int n0   = blockIdx.x * BN;
  const int m0   = blockIdx.y * BM;
  const int tid  = threadIdx.x;
  const int lane = tid & 31;
  const int wid  = tid >> 5;
  const int wm   = (wid & 3) * 64;
  const int wn   = (wid >> 2) * 32;
  const int ln15 = lane & 15;
  const int hi   = (lane >> 4) * 8;
  const int rhi  = (lane >> 4) * 8;

  v8f acc[4][2];
  const v8f zero = {0.f, 0.f, 0.f, 0.f, 0.f, 0.f, 0.f, 0.f};
#pragma unroll
  for (int mt = 0; mt < 4; ++mt)
#pragma unroll
    for (int nt = 0; nt < 2; ++nt) acc[mt][nt] = zero;

  for (int kt = 0; kt < HIDD; kt += BK) {
    __syncthreads();
    // stage A = sigmoid(q) * R, converted to bf16
#pragma unroll
    for (int p = 0; p < 8; ++p) {
      int idx = p * 256 + tid;
      int r = idx >> 3, c4 = (idx & 7) * 4;
      const int gr = m0 + r;
      const int bI = gr / TT;
      const float4 a  = *reinterpret_cast<const float4*>(
          &q[(size_t)gr * HIDD + kt + c4]);
      const float4 rv = *reinterpret_cast<const float4*>(
          &R[(size_t)bI * HIDD + kt + c4]);
      const float y0 = rv.x / (1.f + __expf(-a.x));
      const float y1 = rv.y / (1.f + __expf(-a.y));
      const float y2 = rv.z / (1.f + __expf(-a.z));
      const float y3 = rv.w / (1.f + __expf(-a.w));
      unsigned long long pk =  (unsigned long long)f2bf(y0)
        | ((unsigned long long)f2bf(y1) << 16)
        | ((unsigned long long)f2bf(y2) << 32)
        | ((unsigned long long)f2bf(y3) << 48);
      *reinterpret_cast<unsigned long long*>(&Al[r * LDK + c4]) = pk;
    }
    // stage Wo tile transposed
#pragma unroll
    for (int p = 0; p < 2; ++p) {
      int idx = p * 256 + tid;
      int kr = idx >> 4, c4 = (idx & 15) * 4;
      const float4 b = *reinterpret_cast<const float4*>(
          &Wo[(size_t)(kt + kr) * CC + n0 + c4]);
      Bl[(c4 + 0) * LDK + kr] = f2bf(b.x);
      Bl[(c4 + 1) * LDK + kr] = f2bf(b.y);
      Bl[(c4 + 2) * LDK + kr] = f2bf(b.z);
      Bl[(c4 + 3) * LDK + kr] = f2bf(b.w);
    }
    __syncthreads();

    v16bf af[4], bf[2];
#pragma unroll
    for (int mt = 0; mt < 4; ++mt) af[mt] = load_frag(Al, wm + mt * 16 + ln15, hi);
#pragma unroll
    for (int nt = 0; nt < 2; ++nt) bf[nt] = load_frag(Bl, wn + nt * 16 + ln15, hi);
#pragma unroll
    for (int mt = 0; mt < 4; ++mt)
#pragma unroll
      for (int nt = 0; nt < 2; ++nt)
        acc[mt][nt] = __builtin_amdgcn_wmma_f32_16x16x32_bf16(
            false, af[mt], false, bf[nt], (short)0, acc[mt][nt], false, false);
  }

#pragma unroll
  for (int mt = 0; mt < 4; ++mt)
#pragma unroll
    for (int nt = 0; nt < 2; ++nt) {
      const int col  = n0 + wn + nt * 16 + ln15;
      const float bb = bo[col];
      const int rb   = m0 + wm + mt * 16 + rhi;
#pragma unroll
      for (int r = 0; r < 8; ++r)
        out[(size_t)(rb + r) * CC + col] = acc[mt][nt][r] + bb;
    }
}

// ---------------------------------------------------------------------------
extern "C" void kernel_launch(void* const* d_in, const int* in_sizes, int n_in,
                              void* d_out, int out_size, void* d_ws, size_t ws_size,
                              hipStream_t stream) {
  (void)in_sizes; (void)n_in; (void)out_size; (void)ws_size;
  const float* x  = (const float*)d_in[0];
  const float* Wq = (const float*)d_in[1];
  const float* bq = (const float*)d_in[2];
  const float* Wk = (const float*)d_in[3];
  const float* bk = (const float*)d_in[4];
  const float* Wv = (const float*)d_in[5];
  const float* bv = (const float*)d_in[6];
  const float* Wo = (const float*)d_in[7];
  const float* bo = (const float*)d_in[8];
  // d_in[9] = u, d_in[10] = vp: unused — exp(w_bias - max(axis=0,singleton)) == 1 exactly.

  float* out = (float*)d_out;
  float* ws  = (float*)d_ws;
  const size_t QS = (size_t)MROWS * HIDD;   // 21,233,664 floats each
  float* q  = ws;
  float* k  = q + QS;
  float* v  = k + QS;
  float* mk = v + QS;                       // [T, HID]
  float* R  = mk + (size_t)TT * HIDD;       // [B, HID]

  dim3 blk(256);
  aft_gemm_qkv<<<dim3(HIDD / BN, MROWS / BM, 3), blk, 0, stream>>>(
      x, Wq, Wk, Wv, bq, bk, bv, q, k, v);
  aft_maxk<<<dim3((TT * HIDD) / 256), blk, 0, stream>>>(k, mk);
  aft_sums<<<dim3((BATCH * HIDD) / 256), blk, 0, stream>>>(k, v, mk, R);
  aft_gemm_out<<<dim3(CC / BN, MROWS / BM, 1), blk, 0, stream>>>(q, R, Wo, bo, out);
}
